// PhonemeBSQQuantizer_37666863186438
// MI455X (gfx1250) — matrix operand order repaired and checked by reading, not compiled, not run on hardware
//
#include <hip/hip_runtime.h>

typedef __attribute__((ext_vector_type(16))) __bf16 v16bf;
typedef __attribute__((ext_vector_type(8)))  __bf16 bf16x8;
typedef __attribute__((ext_vector_type(8)))  float  v8f;
typedef __attribute__((ext_vector_type(4)))  float  f32x4;

#define D_DIM 512   // feature dim
#define K_ENT 512   // codebook entries
#define S_DIM 64    // BSQ dim
#define M_TOT 16384 // B*T

#define LDB  520    // padded LDS row stride (elements) for 512-wide rows
#define LDR  72     // padded LDS row stride (elements) for 64-wide rows

// ---------------- CDNA5 async-copy helpers ----------------

__device__ inline unsigned lds_lo32(const void* p) {
    // LDS aperture is 4GB-aligned (ISA aperture check): low 32 bits = LDS offset
    return (unsigned)(size_t)p;
}

__device__ inline void async_copy16(unsigned lds_off, const void* gsrc) {
    // 16-byte global -> LDS async copy, tracked with ASYNCcnt
    asm volatile("global_load_async_to_lds_b128 %0, %1, off"
                 :: "v"(lds_off), "v"(gsrc) : "memory");
}

__device__ inline void wait_async0() {
    asm volatile("s_wait_asynccnt 0x0" ::: "memory");
}

// ---------------- fragment helpers (wave32 WMMA bf16 layouts) ----------------

__device__ inline v16bf make_frag(bf16x8 lo, bf16x8 hi) {
    return __builtin_shufflevector(lo, hi, 0,1,2,3,4,5,6,7,8,9,10,11,12,13,14,15);
}

// A-matrix 16x32 bf16 fragment from an fp32 row (row-major, this lane's row).
// lane<16 holds K{kb+0..7, kb+16..23}; lane>=16 holds K{kb+8..15, kb+24..31}.
__device__ inline v16bf load_afrag_f32(const float* row_ptr, int kb, int half) {
    int klo = kb + half * 8;
    v16bf f;
#pragma unroll
    for (int g = 0; g < 2; ++g) {
        int k0 = klo + g * 16;
        f32x4 a = *(const f32x4*)(row_ptr + k0);
        f32x4 b = *(const f32x4*)(row_ptr + k0 + 4);
#pragma unroll
        for (int i = 0; i < 4; ++i) {
            f[g * 8 + i]     = (__bf16)a[i];
            f[g * 8 + 4 + i] = (__bf16)b[i];
        }
    }
    return f;
}

// Same A layout but source = x - codebook[idx] (residual), fp32 sources.
__device__ inline v16bf load_resfrag(const float* xr, const float* cr, int kb, int half) {
    int klo = kb + half * 8;
    v16bf f;
#pragma unroll
    for (int g = 0; g < 2; ++g) {
        int k0 = klo + g * 16;
        f32x4 a  = *(const f32x4*)(xr + k0);
        f32x4 b  = *(const f32x4*)(xr + k0 + 4);
        f32x4 ca = *(const f32x4*)(cr + k0);
        f32x4 cb = *(const f32x4*)(cr + k0 + 4);
#pragma unroll
        for (int i = 0; i < 4; ++i) {
            f[g * 8 + i]     = (__bf16)(a[i] - ca[i]);
            f[g * 8 + 4 + i] = (__bf16)(b[i] - cb[i]);
        }
    }
    return f;
}

// A layout from a bf16 row-major row (used for the +/-1 Q matrix, ld = 64).
__device__ inline v16bf load_afrag_bf16(const __bf16* row_ptr, int kb, int half) {
    int klo = kb + half * 8;
    bf16x8 lo = *(const bf16x8*)(row_ptr + klo);
    bf16x8 hi = *(const bf16x8*)(row_ptr + klo + 16);
    return make_frag(lo, hi);
}

// B-matrix 32x16 bf16 fragment from LDS: p points at this lane's (row n, k=kb+half*8).
// lanes 0-15 hold K{kb+0..7, kb+16..23}; lanes 16-31 hold K{kb+8..15, kb+24..31}.
__device__ inline v16bf load_bfrag_lds(const __bf16* p) {
    bf16x8 lo = *(const bf16x8*)(p);
    bf16x8 hi = *(const bf16x8*)(p + 16);
    return make_frag(lo, hi);
}

__device__ inline v8f wmma_bf16(v16bf a, v16bf b, v8f c) {
    return __builtin_amdgcn_wmma_f32_16x16x32_bf16(false, a, false, b, (short)0, c, false, false);
}

// ---------------- kernel 1: precompute bf16 operands + ||c||^2 ----------------

__global__ __launch_bounds__(256) void prep_kernel(const float* __restrict__ cb,
                                                   const float* __restrict__ Wp,
                                                   const float* __restrict__ Wr,
                                                   __bf16* __restrict__ cbB,
                                                   __bf16* __restrict__ WpT,
                                                   __bf16* __restrict__ WrT,
                                                   float* __restrict__ csq) {
    int i = blockIdx.x * 256 + threadIdx.x;
    if (i < K_ENT * D_DIM) { cbB[i] = (__bf16)cb[i]; return; }
    int j = i - K_ENT * D_DIM;
    if (j < S_DIM * D_DIM) {               // WpT[s][d] = Wp[d][s]
        int s = j >> 9, d = j & 511;
        WpT[j] = (__bf16)Wp[d * S_DIM + s];
        return;
    }
    int k = j - S_DIM * D_DIM;
    if (k < D_DIM * S_DIM) {               // WrT[dd][s] = Wr[s][dd]
        int dd = k >> 6, s = k & 63;
        WrT[k] = (__bf16)Wr[s * D_DIM + dd];
        return;
    }
    int e = k - D_DIM * S_DIM;
    if (e < K_ENT) {
        const float* r = cb + e * D_DIM;
        float sum = 0.f;
        for (int d = 0; d < D_DIM; ++d) sum += r[d] * r[d];
        csq[e] = sum;
    }
}

// ---------------- kernel 2: VQ distances via WMMA + argmin ----------------
// One wave handles a 16-row tile of x against all 512 codebook entries.
// Codebook tiles (16 entries x 512 dims, bf16) are double-buffered in LDS via
// async global->LDS copies; stage of tile nt+1 overlaps the 16 WMMAs of tile nt.

__global__ __launch_bounds__(256) void vq_kernel(const float* __restrict__ x,
                                                 const __bf16* __restrict__ cbB,
                                                 const float* __restrict__ csq,
                                                 int* __restrict__ idx_ws,
                                                 float* __restrict__ idx_out) {
    __shared__ __bf16 ldsb[2][16 * LDB];   // 2 x 16.25 KB, padded rows
    int tid = threadIdx.x;
    int wid = tid >> 5, lane = tid & 31;
    int half = lane >> 4, lr = lane & 15;
    int m0 = (blockIdx.x * 8 + wid) * 16;
    unsigned lbase = lds_lo32(&ldsb[0][0]);

    // prologue: stage codebook tile 0 into buffer 0 (1024 x 16B chunks, 4/thread)
    {
        const char* g = (const char*)cbB;
#pragma unroll
        for (int r = 0; r < 4; ++r) {
            int chunk = tid + r * 256;
            int e = chunk >> 6, w = chunk & 63;
            async_copy16(lbase + (unsigned)(e * (LDB * 2) + w * 16), g + chunk * 16);
        }
    }

    // A fragments: this wave's 16 x-rows, K=512 (overlaps the async stage)
    const float* xrow = x + (size_t)(m0 + lr) * D_DIM;
    v16bf afr[16];
#pragma unroll
    for (int t = 0; t < 16; ++t) afr[t] = load_afrag_f32(xrow, t * 32, half);

    float minv[8];
    int   mini[8];
#pragma unroll
    for (int j = 0; j < 8; ++j) { minv[j] = 3.4e38f; mini[j] = 0; }

    for (int nt = 0; nt < 32; ++nt) {
        wait_async0();        // my async writes for tile nt are done
        __syncthreads();      // everyone's are -> buffer nt&1 is valid, buffer (nt+1)&1 is free
        if (nt + 1 < 32) {    // stage next tile into the other buffer (overlaps WMMAs below)
            const char* g = (const char*)(cbB + (size_t)(nt + 1) * 16 * D_DIM);
            unsigned dstb = lbase + (unsigned)(((nt + 1) & 1) * (16 * LDB * 2));
#pragma unroll
            for (int r = 0; r < 4; ++r) {
                int chunk = tid + r * 256;
                int e = chunk >> 6, w = chunk & 63;
                async_copy16(dstb + (unsigned)(e * (LDB * 2) + w * 16), g + chunk * 16);
            }
        }
        const __bf16* bt = &ldsb[nt & 1][0] + lr * LDB + half * 8;
        v8f acc = {};
#pragma unroll
        for (int t = 0; t < 16; ++t) {
            acc = wmma_bf16(afr[t], load_bfrag_lds(bt + t * 32), acc);
        }
        int n = nt * 16 + lr;
        float cs = csq[n];
#pragma unroll
        for (int j = 0; j < 8; ++j) {
            float dist = cs - 2.0f * acc[j];   // z_sq is row-constant, dropped
            bool lt = dist < minv[j];
            minv[j] = lt ? dist : minv[j];
            mini[j] = lt ? n : mini[j];
        }
    }
    // reduce across the 16 lanes of each half (columns); rows stay per-half
#pragma unroll
    for (int m = 1; m < 16; m <<= 1) {
#pragma unroll
        for (int j = 0; j < 8; ++j) {
            float ov = __shfl_xor(minv[j], m, 32);
            int   oi = __shfl_xor(mini[j], m, 32);
            bool take = (ov < minv[j]) || (ov == minv[j] && oi < mini[j]);
            minv[j] = take ? ov : minv[j];
            mini[j] = take ? oi : mini[j];
        }
    }
    if (lr == 0) {   // lanes 0 and 16 write rows 0..7 / 8..15
#pragma unroll
        for (int j = 0; j < 8; ++j) {
            int row = m0 + half * 8 + j;
            idx_ws[row]  = mini[j];
            idx_out[row] = (float)mini[j];
        }
    }
}

// ---------------- kernel 3: residual @ Wp -> codes (+/-1 Q matrix) ----------------
// Whole WpT (64 x 512 bf16) staged once into LDS via async copies.

__global__ __launch_bounds__(256) void bsq_kernel(const float* __restrict__ x,
                                                  const float* __restrict__ cb,
                                                  const __bf16* __restrict__ WpT,
                                                  const float* __restrict__ bp,
                                                  const int* __restrict__ idx_ws,
                                                  float* __restrict__ codes_out,
                                                  __bf16* __restrict__ Q) {
    __shared__ __bf16 ldw[S_DIM * LDB];    // 65 KB padded
    int tid = threadIdx.x;
    int wid = tid >> 5, lane = tid & 31;
    int half = lane >> 4, lr = lane & 15;
    int m0 = (blockIdx.x * 8 + wid) * 16;
    unsigned lbase = lds_lo32(ldw);

    // stage WpT: 64 rows x 64 chunks of 16B = 4096 chunks, 16/thread
#pragma unroll
    for (int r = 0; r < 16; ++r) {
        int chunk = tid + r * 256;
        int row = chunk >> 6, w = chunk & 63;
        async_copy16(lbase + (unsigned)(row * (LDB * 2) + w * 16),
                     (const char*)WpT + chunk * 16);
    }

    // residual A fragments (overlaps the async stage)
    int myrow = m0 + lr;
    const float* xrow = x + (size_t)myrow * D_DIM;
    const float* crow = cb + (size_t)idx_ws[myrow] * D_DIM;
    v16bf afr[16];
#pragma unroll
    for (int t = 0; t < 16; ++t) afr[t] = load_resfrag(xrow, crow, t * 32, half);

    wait_async0();
    __syncthreads();

#pragma unroll
    for (int nt = 0; nt < 4; ++nt) {
        v8f acc = {};
        int s = nt * 16 + lr;
        const __bf16* bt = ldw + s * LDB + half * 8;
#pragma unroll
        for (int t = 0; t < 16; ++t) {
            acc = wmma_bf16(afr[t], load_bfrag_lds(bt + t * 32), acc);
        }
        float bias = bp[s];
#pragma unroll
        for (int j = 0; j < 8; ++j) {
            // sign(spherical) == sign(spherical / ||.||): skip the norm entirely
            float code = (acc[j] + bias) > 0.0f ? 1.0f : 0.0f;
            int row = m0 + half * 8 + j;
            codes_out[(size_t)row * S_DIM + s] = code;
            Q[(size_t)row * S_DIM + s] = (__bf16)(2.0f * code - 1.0f);
        }
    }
}

// ---------------- kernel 4: quantized @ Wr + br + z_q -> reconstructed ----------------
// Whole WrT (512 x 64 bf16) staged once into LDS via async copies.

__global__ __launch_bounds__(256) void rec_kernel(const float* __restrict__ cb,
                                                  const __bf16* __restrict__ WrT,
                                                  const float* __restrict__ br,
                                                  const int* __restrict__ idx_ws,
                                                  const __bf16* __restrict__ Q,
                                                  float* __restrict__ rec_out) {
    __shared__ __bf16 ldr[D_DIM * LDR];    // 72 KB padded
    int tid = threadIdx.x;
    int wid = tid >> 5, lane = tid & 31;
    int half = lane >> 4, lr = lane & 15;
    int m0 = (blockIdx.x * 8 + wid) * 16;
    unsigned lbase = lds_lo32(ldr);

    // stage WrT: 512 rows x 8 chunks of 16B = 4096 chunks, 16/thread
#pragma unroll
    for (int r = 0; r < 16; ++r) {
        int chunk = tid + r * 256;
        int row = chunk >> 3, w = chunk & 7;
        async_copy16(lbase + (unsigned)(row * (LDR * 2) + w * 16),
                     (const char*)WrT + chunk * 16);
    }

    // A fragments from the +/-1 matrix and z_q row indices (overlap the stage)
    const __bf16* qrow = Q + (size_t)(m0 + lr) * S_DIM;
    v16bf a0 = load_afrag_bf16(qrow, 0, half);
    v16bf a1 = load_afrag_bf16(qrow, 32, half);
    int idx8[8];
#pragma unroll
    for (int j = 0; j < 8; ++j) idx8[j] = idx_ws[m0 + half * 8 + j];

    wait_async0();
    __syncthreads();

    for (int nt = 0; nt < 32; ++nt) {
        int dd = nt * 16 + lr;
        const __bf16* bt = ldr + dd * LDR + half * 8;
        v8f acc = {};
        acc = wmma_bf16(a0, load_bfrag_lds(bt), acc);
        acc = wmma_bf16(a1, load_bfrag_lds(bt + 32), acc);
        float bias = br[dd];
#pragma unroll
        for (int j = 0; j < 8; ++j) {
            float zq = cb[(size_t)idx8[j] * D_DIM + dd];   // reconstructed = z_q + acoustic
            int row = m0 + half * 8 + j;
            rec_out[(size_t)row * D_DIM + dd] = acc[j] + bias + zq;
        }
    }
}

// ---------------- host launch ----------------

extern "C" void kernel_launch(void* const* d_in, const int* in_sizes, int n_in,
                              void* d_out, int out_size, void* d_ws, size_t ws_size,
                              hipStream_t stream) {
    (void)in_sizes; (void)n_in; (void)out_size; (void)ws_size;
    const float* x  = (const float*)d_in[0];
    const float* cb = (const float*)d_in[1];
    const float* Wp = (const float*)d_in[2];
    const float* bp = (const float*)d_in[3];
    const float* Wr = (const float*)d_in[4];
    const float* br = (const float*)d_in[5];

    // workspace layout (bytes)
    char* ws = (char*)d_ws;
    __bf16* cbB   = (__bf16*)(ws + 0);            // 512*512*2  = 524288
    __bf16* WpT   = (__bf16*)(ws + 524288);       // 64*512*2   = 65536
    __bf16* WrT   = (__bf16*)(ws + 589824);       // 512*64*2   = 65536
    float*  csq   = (float*) (ws + 655360);       // 512*4      = 2048
    int*    idxws = (int*)   (ws + 657408);       // 16384*4    = 65536
    __bf16* Q     = (__bf16*)(ws + 722944);       // 16384*64*2 = 2097152 -> total ~2.8MB

    // d_out: [reconstructed 16384*512][indices 16384][codes 16384*64], all fp32
    float* rec_out   = (float*)d_out;
    float* idx_out   = rec_out + (size_t)M_TOT * D_DIM;
    float* codes_out = idx_out + M_TOT;

    const int PREP_N = K_ENT * D_DIM + S_DIM * D_DIM + D_DIM * S_DIM + K_ENT;
    prep_kernel<<<(PREP_N + 255) / 256, 256, 0, stream>>>(cb, Wp, Wr, cbB, WpT, WrT, csq);

    // 16384 rows / 16 per wave / 8 waves per block = 128 blocks
    vq_kernel <<<128, 256, 0, stream>>>(x, cbB, csq, idxws, idx_out);
    bsq_kernel<<<128, 256, 0, stream>>>(x, cb, WpT, bp, idxws, codes_out, Q);
    rec_kernel<<<128, 256, 0, stream>>>(cb, WrT, br, idxws, Q, rec_out);
}
